// DurIAN_75703093559821
// MI455X (gfx1250) — matrix operand order repaired and checked by reading, not compiled, not run on hardware
//
#include <hip/hip_runtime.h>
#include <cstdint>

// =====================================================================
// DurIAN forward for MI455X (gfx1250, wave32).
//  - All batched matmuls -> v_wmma_f32_16x16x32_f16, 16x16 C tile per wave,
//    4 waves/block sharing one N-tile; B tile staged through LDS.
//  - Weights pre-packed per 32-K chunk so fragment loads are contiguous b128s.
//  - Convolutions = per-tap accumulating WMMA GEMMs over row-padded f16
//    activation buffers (no im2col materialization).
//  - BN/bias/ReLU/sigmoid folded into GEMM epilogue.
//  - Sequential GRU scans = persistent single-workgroup kernels, state in LDS.
// =====================================================================

typedef __attribute__((ext_vector_type(16))) _Float16 v16h;
typedef __attribute__((ext_vector_type(8)))  _Float16 v8h;
typedef __attribute__((ext_vector_type(8)))  float    v8f;

#define TPHN  2048
#define TMEL  16384
#define EDIM  256
#define HENC  128
#define DDEC  512
#define MMEL  80
#define CNT   (TMEL / 4)   // 4096 decoder steps

__device__ __forceinline__ float sigm(float x) { return 1.f / (1.f + __expf(-x)); }

__device__ __forceinline__ v16h join8(v8h lo, v8h hi) {
  return __builtin_shufflevector(lo, hi, 0,1,2,3,4,5,6,7,8,9,10,11,12,13,14,15);
}

// ------------------------------------------------------------------
// WMMA GEMM: C[M,N] (+)= act( A[MxK]f16 * Bp * scale[n] + shift[n] )
// Bp is chunk-packed: element (k,n) at ((k/32)*NP + n)*32 + (k%32).
// flags bit0 = accumulate into C, bit1 = apply epilogue.
// Requires: M%64==0, N%16==0 (N==NP), K%32==0, K>=32.
// ------------------------------------------------------------------
__global__ __launch_bounds__(128) void wmma_gemm_kernel(
    const _Float16* __restrict__ A, int lda,
    const _Float16* __restrict__ Bp, int NP,
    float* __restrict__ C, int ldc,
    const float* __restrict__ scale, const float* __restrict__ shift,
    int M, int N, int K, int act, int flags)
{
  __shared__ _Float16 lb[64 * 16];            // one 64K x 16N B tile (2 KB)
  const int tid  = threadIdx.x;
  const int wave = tid >> 5;
  const int lane = tid & 31;
  const int m0 = (blockIdx.y * 4 + wave) * 16;
  const int n0 = blockIdx.x * 16;
  const int half  = lane >> 4;                // 0: lanes 0-15, 1: lanes 16-31
  const int l15   = lane & 15;
  const int abase = half * 16;                // A K-offset base within 32-chunk
  const int bbase = half * 16;                // B K-offset base within 32-chunk
  const _Float16* arow = A + (size_t)(m0 + l15) * lda;
  const int swhich = tid >> 6;                // staging: which 32-chunk
  const int selem  = (tid & 63) * 8;          // staging: 8 halves per thread

  v8f acc = {};
  int k0 = 0;
  for (; k0 + 64 <= K; k0 += 64) {
    // cooperative stage: two 32-K chunks of the 16-column B tile -> LDS
    {
      const int c = (k0 >> 5) + swhich;
      const v8h bv = *(const v8h*)(Bp + ((size_t)c * NP + n0) * 32 + selem);
      *(v8h*)(lb + swhich * 512 + selem) = bv;
    }
    __builtin_prefetch(arow + k0 + 128, 0, 1);    // global_prefetch_b8 on A
    __syncthreads();
    const v16h a0 = join8(*(const v8h*)(arow + k0 + abase),
                          *(const v8h*)(arow + k0 + abase + 8));
    const v16h a1 = join8(*(const v8h*)(arow + k0 + 32 + abase),
                          *(const v8h*)(arow + k0 + 32 + abase + 8));
    const v16h b0 = *(const v16h*)(lb + l15 * 32 + bbase);
    const v16h b1 = *(const v16h*)(lb + 512 + l15 * 32 + bbase);
    acc = __builtin_amdgcn_wmma_f32_16x16x32_f16(false, a0, false, b0,
                                                 (short)0, acc, false, false);
    acc = __builtin_amdgcn_wmma_f32_16x16x32_f16(false, a1, false, b1,
                                                 (short)0, acc, false, false);
    __syncthreads();
  }
  if (k0 < K) {                                // 32-K tail
    if (tid < 64) {
      const int c = k0 >> 5;
      const v8h bv = *(const v8h*)(Bp + ((size_t)c * NP + n0) * 32 + selem);
      *(v8h*)(lb + selem) = bv;
    }
    __syncthreads();
    const v16h a0 = join8(*(const v8h*)(arow + k0 + abase),
                          *(const v8h*)(arow + k0 + abase + 8));
    const v16h b0 = *(const v16h*)(lb + l15 * 32 + bbase);
    acc = __builtin_amdgcn_wmma_f32_16x16x32_f16(false, a0, false, b0,
                                                 (short)0, acc, false, false);
  }

  const int col = n0 + l15;
#pragma unroll
  for (int i = 0; i < 8; ++i) {
    const int row = m0 + half * 8 + i;        // ISA C layout: VGPR i -> M=i / 8+i
    const size_t idx = (size_t)row * ldc + col;
    float v = acc[i];
    if (flags & 1) v += C[idx];
    if (flags & 2) {
      const float s  = scale ? scale[col] : 1.f;
      const float sh = shift ? shift[col] : 0.f;
      v = v * s + sh;
      if (act == 1)      v = fmaxf(v, 0.f);
      else if (act == 2) v = sigm(v);
    }
    C[idx] = v;
  }
}

// ---------------- persistent bidirectional GRU (block 0 = fwd, 1 = bwd) ----
__global__ __launch_bounds__(384) void gru_bidir_kernel(
    const float* __restrict__ gi_f, const float* __restrict__ gi_b,
    const float* __restrict__ whh_f, const float* __restrict__ whh_b,
    const float* __restrict__ bhh_f, const float* __restrict__ bhh_b,
    float* __restrict__ out, int T, int H, int OH)
{
  __shared__ float sh[HENC];       // H <= 128
  __shared__ float sg[3 * HENC];   // 3H <= 384
  const int dir = blockIdx.x;
  const float* gi  = dir ? gi_b  : gi_f;
  const float* whh = dir ? whh_b : whh_f;
  const float* bhh = dir ? bhh_b : bhh_f;
  const int n = threadIdx.x;       // blockDim.x == 3H
  const int off = dir * H;
  if (n < H) sh[n] = 0.f;
  __syncthreads();
  for (int s = 0; s < T; ++s) {
    const int idx = dir ? (T - 1 - s) : s;
    const float* w = whh + (size_t)n * H;
    float a = bhh[n];
    for (int k = 0; k < H; ++k) a += w[k] * sh[k];
    sg[n] = a;
    __syncthreads();
    if (n < H) {
      const float* g = gi + (size_t)idx * 3 * H;
      const float r  = sigm(g[n]       + sg[n]);
      const float z  = sigm(g[H + n]   + sg[H + n]);
      const float nn = tanhf(g[2*H + n] + r * sg[2*H + n]);
      const float hn = (1.f - z) * nn + z * sh[n];
      out[(size_t)idx * OH + off + n] = hn;
      sh[n] = hn;
    }
    __syncthreads();
  }
}

// ---------------- persistent decoder: prenet + 2x GRU(512) + mel linear ----
__global__ __launch_bounds__(512) void decoder_kernel(
    const float* __restrict__ gimem,   // [CNT x 1536] mem-part gates + b_ih0
    const float* __restrict__ w0ih,    // [1536 x 512] (cols 0..255 = prenet)
    const float* __restrict__ w0hh, const float* __restrict__ b0hh,
    const float* __restrict__ w1ih, const float* __restrict__ w1hh,
    const float* __restrict__ b1ih, const float* __restrict__ b1hh,
    const float* __restrict__ pW1, const float* __restrict__ pb1,
    const float* __restrict__ pW2, const float* __restrict__ pb2,
    const float* __restrict__ mlW, const float* __restrict__ mlb,
    float* __restrict__ mel, int steps)
{
  __shared__ float h0[512], h1[512], p1[256], p2[256], prev[80];
  __shared__ float sgi[1536], sgh[1536];
  const int tid = threadIdx.x;
  h0[tid] = 0.f; h1[tid] = 0.f;
  if (tid < 80) prev[tid] = 0.f;
  __syncthreads();
  for (int t = 0; t < steps; ++t) {
    if (tid < 256) {                       // prenet layer 1 (80 -> 256) + relu
      float a = pb1[tid];
      const float* w = pW1 + (size_t)tid * 80;
      for (int k = 0; k < 80; ++k) a += w[k] * prev[k];
      p1[tid] = fmaxf(a, 0.f);
    }
    __syncthreads();
    if (tid < 256) {                       // prenet layer 2 (256 -> 256) + relu
      float a = pb2[tid];
      const float* w = pW2 + (size_t)tid * 256;
      for (int k = 0; k < 256; ++k) a += w[k] * p1[k];
      p2[tid] = fmaxf(a, 0.f);
    }
    __syncthreads();
#pragma unroll
    for (int j = 0; j < 3; ++j) {          // dec0 gates
      const int n = tid + j * 512;
      float gi = gimem[(size_t)t * 1536 + n];
      const float* wi = w0ih + (size_t)n * 512;      // prenet half of W_ih
      for (int k = 0; k < 256; ++k) gi += wi[k] * p2[k];
      float gh = b0hh[n];
      const float* wh = w0hh + (size_t)n * 512;
      for (int k = 0; k < 512; ++k) gh += wh[k] * h0[k];
      sgi[n] = gi; sgh[n] = gh;
    }
    __syncthreads();
    {
      const float r  = sigm(sgi[tid]        + sgh[tid]);
      const float z  = sigm(sgi[512 + tid]  + sgh[512 + tid]);
      const float nn = tanhf(sgi[1024 + tid] + r * sgh[1024 + tid]);
      h0[tid] = (1.f - z) * nn + z * h0[tid];
    }
    __syncthreads();
#pragma unroll
    for (int j = 0; j < 3; ++j) {          // dec1 gates
      const int n = tid + j * 512;
      float gi = b1ih[n];
      const float* wi = w1ih + (size_t)n * 512;
      for (int k = 0; k < 512; ++k) gi += wi[k] * h0[k];
      float gh = b1hh[n];
      const float* wh = w1hh + (size_t)n * 512;
      for (int k = 0; k < 512; ++k) gh += wh[k] * h1[k];
      sgi[n] = gi; sgh[n] = gh;
    }
    __syncthreads();
    {
      const float r  = sigm(sgi[tid]        + sgh[tid]);
      const float z  = sigm(sgi[512 + tid]  + sgh[512 + tid]);
      const float nn = tanhf(sgi[1024 + tid] + r * sgh[1024 + tid]);
      h1[tid] = (1.f - z) * nn + z * h1[tid];
    }
    __syncthreads();
    if (tid < 320) {                       // mel linear (512 -> 4x80)
      float a = mlb[tid];
      const float* w = mlW + (size_t)tid * 512;
      for (int k = 0; k < 512; ++k) a += w[k] * h1[k];
      mel[(size_t)t * 320 + tid] = a;
      if (tid >= 240) prev[tid - 240] = a; // last frame feeds prenet
    }
    __syncthreads();
  }
}

// ---------------- small helper kernels ------------------------------------
__global__ void embed_kernel(const int* __restrict__ src,
                             const float* __restrict__ emb,
                             float* __restrict__ x)
{
  const int t = blockIdx.x, e = threadIdx.x;
  x[(size_t)t * EDIM + e] = emb[(size_t)src[t] * EDIM + e];
}

__global__ void cumsum_kernel(const int* __restrict__ d, int* __restrict__ c, int n)
{
  if (blockIdx.x == 0 && threadIdx.x == 0) {
    int s = 0;
    for (int i = 0; i < n; ++i) { s += d[i]; c[i] = s; }
  }
}

__global__ void pool_memory_kernel(const int* __restrict__ csum,
                                   const float* __restrict__ enc,
                                   float* __restrict__ mp)
{
  __shared__ int sidx[4];
  const int c = blockIdx.x;
  if (threadIdx.x < 4) {
    const int tm = c * 4 + threadIdx.x;
    int lo = 0, hi = TPHN;                 // searchsorted(csum, tm, 'right')
    while (lo < hi) { int mid = (lo + hi) >> 1; if (csum[mid] <= tm) lo = mid + 1; else hi = mid; }
    sidx[threadIdx.x] = min(lo, TPHN - 1);
  }
  __syncthreads();
  const int e = threadIdx.x;
  float a = 0.f;
  for (int s = 0; s < 4; ++s) a += enc[(size_t)sidx[s] * EDIM + e];
  mp[(size_t)c * EDIM + e] = a * 0.25f;
}

// f32 [MxN] -> f16 [MPxNP], row-padded by `rowpad`, zero-filled outside.
__global__ void cvt_pad_f16_kernel(const float* __restrict__ src, _Float16* __restrict__ dst,
                                   int M, int N, int MP, int NP, int rowpad)
{
  const size_t i = (size_t)blockIdx.x * 256 + threadIdx.x;
  if (i >= (size_t)MP * NP) return;
  const int r = (int)(i / NP), c = (int)(i % NP);
  const int sr = r - rowpad;
  float v = (sr >= 0 && sr < M && c < N) ? src[(size_t)sr * N + c] : 0.f;
  dst[i] = (_Float16)v;
}

// W[O x I x KW] f32 -> per-tap chunk-packed f16 B: element (dt,k,n) at
// dt*KP*NP + ((k/32)*NP + n)*32 + (k%32).  (KW==1: plain W^T, packed.)
__global__ void wtap_f16_kernel(const float* __restrict__ W, _Float16* __restrict__ Bt,
                                int O, int I, int KW, int KP, int NP)
{
  const size_t i = (size_t)blockIdx.x * 256 + threadIdx.x;
  if (i >= (size_t)KW * KP * NP) return;
  const int dt = (int)(i / ((size_t)KP * NP));
  const size_t rem = i % ((size_t)KP * NP);
  const int k = (int)(rem / NP), n = (int)(rem % NP);
  float v = (k < I && n < O) ? W[((size_t)n * I + k) * KW + dt] : 0.f;
  const size_t off = (size_t)dt * KP * NP + ((size_t)(k >> 5) * NP + n) * 32 + (k & 31);
  Bt[off] = (_Float16)v;
}

// dec0 W_ih[:, 256:512]^T -> chunk-packed f16 [K=256 x N=1536]
__global__ void wslice_f16_kernel(const float* __restrict__ W, _Float16* __restrict__ Bt)
{
  const size_t i = (size_t)blockIdx.x * 256 + threadIdx.x;
  if (i >= (size_t)256 * 1536) return;
  const int k = (int)(i / 1536), n = (int)(i % 1536);
  const size_t off = ((size_t)(k >> 5) * 1536 + n) * 32 + (k & 31);
  Bt[off] = (_Float16)W[(size_t)n * 512 + 256 + k];
}

__global__ void fold_bn_kernel(const float* __restrict__ b, const float* __restrict__ sc,
                               const float* __restrict__ sh, float* __restrict__ o, int n)
{
  const int i = blockIdx.x * 256 + threadIdx.x;
  if (i < n) o[i] = b[i] * sc[i] + sh[i];
}

// ym[t] = max(y[t], y[t-1] or -1e30), -> f16 with 1 zero row each side.
__global__ void maxpool_cvt_kernel(const float* __restrict__ y, _Float16* __restrict__ dst,
                                   int T, int C)
{
  const size_t i = (size_t)blockIdx.x * 256 + threadIdx.x;
  if (i >= (size_t)(T + 2) * C) return;
  const int r = (int)(i / C), c = (int)(i % C);
  const int t = r - 1;
  float v = 0.f;
  if (t >= 0 && t < T) {
    const float a = y[(size_t)t * C + c];
    const float b = (t > 0) ? y[(size_t)(t - 1) * C + c] : -1e30f;
    v = fmaxf(a, b);
  }
  dst[i] = (_Float16)v;
}

__global__ void add_kernel(const float* __restrict__ a, float* __restrict__ b, long n)
{
  const long i = (long)blockIdx.x * 256 + threadIdx.x;
  if (i < n) b[i] += a[i];
}

__global__ void highway_kernel(const float* __restrict__ hh, const float* __restrict__ ht,
                               float* __restrict__ y, long n)
{
  const long i = (long)blockIdx.x * 256 + threadIdx.x;
  if (i < n) { const float t = ht[i]; y[i] = hh[i] * t + y[i] * (1.f - t); }
}

// =====================================================================
extern "C" void kernel_launch(void* const* d_in, const int* in_sizes, int n_in,
                              void* d_out, int out_size, void* d_ws, size_t ws_size,
                              hipStream_t stream)
{
  (void)in_sizes; (void)n_in; (void)out_size; (void)ws_size;
  // ---- inputs (setup_inputs dict insertion order, params flattened in order)
  const int*   src = (const int*)d_in[0];
  const int*   dur = (const int*)d_in[1];
  const float* emb = (const float*)d_in[2];
  const float *cW[3], *cB[3], *cS[3], *cSh[3];
  for (int i = 0; i < 3; ++i) {
    cW[i]  = (const float*)d_in[3 + 4*i]; cB[i]  = (const float*)d_in[4 + 4*i];
    cS[i]  = (const float*)d_in[5 + 4*i]; cSh[i] = (const float*)d_in[6 + 4*i];
  }
  const float* efWih=(const float*)d_in[15]; const float* efWhh=(const float*)d_in[16];
  const float* efbih=(const float*)d_in[17]; const float* efbhh=(const float*)d_in[18];
  const float* ebWih=(const float*)d_in[19]; const float* ebWhh=(const float*)d_in[20];
  const float* ebbih=(const float*)d_in[21]; const float* ebbhh=(const float*)d_in[22];
  const float* pW1=(const float*)d_in[23]; const float* pb1=(const float*)d_in[24];
  const float* pW2=(const float*)d_in[25]; const float* pb2=(const float*)d_in[26];
  const float* d0Wih=(const float*)d_in[27]; const float* d0Whh=(const float*)d_in[28];
  const float* d0bih=(const float*)d_in[29]; const float* d0bhh=(const float*)d_in[30];
  const float* d1Wih=(const float*)d_in[31]; const float* d1Whh=(const float*)d_in[32];
  const float* d1bih=(const float*)d_in[33]; const float* d1bhh=(const float*)d_in[34];
  const float* mlW=(const float*)d_in[35];  const float* mlb=(const float*)d_in[36];
  // cbhg: banks 37..68, proj1 69..72, proj2 73..76, highways 77..92,
  //       gru_fwd 93..96, gru_bwd 97..100, last 101..102
  const float* q1W=(const float*)d_in[69]; const float* q1b=(const float*)d_in[70];
  const float* q1s=(const float*)d_in[71]; const float* q1h=(const float*)d_in[72];
  const float* q2W=(const float*)d_in[73]; const float* q2b=(const float*)d_in[74];
  const float* q2s=(const float*)d_in[75]; const float* q2h=(const float*)d_in[76];
  const float* gfWih=(const float*)d_in[93]; const float* gfWhh=(const float*)d_in[94];
  const float* gfbih=(const float*)d_in[95]; const float* gfbhh=(const float*)d_in[96];
  const float* gbWih=(const float*)d_in[97]; const float* gbWhh=(const float*)d_in[98];
  const float* gbbih=(const float*)d_in[99]; const float* gbbhh=(const float*)d_in[100];
  const float* llW=(const float*)d_in[101]; const float* llb=(const float*)d_in[102];

  // ---- workspace bump allocator (total ~205 MB)
  char* wsb = (char*)d_ws;
  size_t cur = 0;
  auto alloc = [&](size_t bytes) -> void* {
    void* p = wsb + cur; cur = (cur + bytes + 255) & ~(size_t)255; return p;
  };
  float*    xA     = (float*)alloc(4ull * TPHN * EDIM);
  float*    xB     = (float*)alloc(4ull * TPHN * EDIM);
  _Float16* xpad   = (_Float16*)alloc(2ull * (TPHN + 4) * EDIM);
  _Float16* BtC    = (_Float16*)alloc(2ull * 5 * EDIM * EDIM);
  float*    fbuf   = (float*)alloc(4ull * 2048);
  float*    giF    = (float*)alloc(4ull * TPHN * 3 * HENC);
  float*    giB    = (float*)alloc(4ull * TPHN * 3 * HENC);
  _Float16* BtG    = (_Float16*)alloc(2ull * EDIM * 3 * HENC);
  float*    enc    = (float*)alloc(4ull * TPHN * EDIM);
  int*      csum   = (int*)alloc(4ull * TPHN);
  float*    mp     = (float*)alloc(4ull * CNT * EDIM);
  _Float16* mph    = (_Float16*)alloc(2ull * CNT * EDIM);
  _Float16* BtD    = (_Float16*)alloc(2ull * EDIM * 3 * DDEC);
  float*    gimem  = (float*)alloc(4ull * CNT * 3 * DDEC);
  _Float16* melpad = (_Float16*)alloc(2ull * (TMEL + 8) * 96);
  _Float16* BtBank = (_Float16*)alloc(2ull * 8 * 96 * MMEL);
  float*    ycat   = (float*)alloc(4ull * TMEL * 640);
  _Float16* ypool  = (_Float16*)alloc(2ull * (TMEL + 2) * 640);
  _Float16* BtP1   = (_Float16*)alloc(2ull * 3 * 640 * 256);
  float*    p1b    = (float*)alloc(4ull * TMEL * 256);
  _Float16* p1pad  = (_Float16*)alloc(2ull * (TMEL + 2) * 256);
  _Float16* BtP2   = (_Float16*)alloc(2ull * 3 * 256 * MMEL);
  float*    hwy    = (float*)alloc(4ull * TMEL * MMEL);
  _Float16* hwypad = (_Float16*)alloc(2ull * TMEL * 96);
  _Float16* BtHW   = (_Float16*)alloc(2ull * 96 * MMEL);
  float*    hh     = (float*)alloc(4ull * TMEL * MMEL);
  float*    ht     = (float*)alloc(4ull * TMEL * MMEL);
  _Float16* BtCG   = (_Float16*)alloc(2ull * 96 * 240);
  float*    gicf   = (float*)alloc(4ull * TMEL * 240);
  float*    gicb   = (float*)alloc(4ull * TMEL * 240);
  float*    cb     = (float*)alloc(4ull * TMEL * 160);
  _Float16* cbh    = (_Float16*)alloc(2ull * TMEL * 160);
  _Float16* BtLL   = (_Float16*)alloc(2ull * 160 * MMEL);

  // ---- launch helpers (all on `stream`; stream order makes buffer reuse safe)
  auto gemm = [&](const _Float16* A, int lda, const _Float16* B, int NP,
                  float* C, int ldc, const float* sc, const float* sh,
                  int M, int N, int K, int act, int flags) {
    dim3 g((unsigned)(N / 16), (unsigned)(M / 64));
    wmma_gemm_kernel<<<g, dim3(128), 0, stream>>>(A, lda, B, NP, C, ldc, sc, sh, M, N, K, act, flags);
  };
  auto cvt = [&](const float* s, _Float16* d, int M, int N, int MP, int NP, int rp) {
    size_t tot = (size_t)MP * NP;
    cvt_pad_f16_kernel<<<dim3((unsigned)((tot + 255) / 256)), dim3(256), 0, stream>>>(s, d, M, N, MP, NP, rp);
  };
  auto wtap = [&](const float* W, _Float16* Bt, int O, int I, int KW, int KP, int NP) {
    size_t tot = (size_t)KW * KP * NP;
    wtap_f16_kernel<<<dim3((unsigned)((tot + 255) / 256)), dim3(256), 0, stream>>>(W, Bt, O, I, KW, KP, NP);
  };
  auto fold = [&](const float* b, const float* sc, const float* sh, int n) {
    fold_bn_kernel<<<dim3((unsigned)((n + 255) / 256)), dim3(256), 0, stream>>>(b, sc, sh, fbuf, n);
  };
  auto conv_taps = [&](const _Float16* Apad, int lda, int rowpad, int p, int KW,
                       const _Float16* Bt, int KP, int NPld,
                       float* C, int ldc, const float* sc, const float* sh,
                       int M, int N, int K, int act) {
    for (int dt = 0; dt < KW; ++dt) {
      int fl = (dt == 0 ? 0 : 1) | (dt == KW - 1 ? 2 : 0);
      gemm(Apad + (size_t)(rowpad + dt - p) * lda, lda,
           Bt + (size_t)dt * KP * NPld, NPld, C, ldc, sc, sh, M, N, K, act, fl);
    }
  };
  auto addv = [&](const float* a, float* b, long n) {
    add_kernel<<<dim3((unsigned)((n + 255) / 256)), dim3(256), 0, stream>>>(a, b, n);
  };

  float* melout = (float*)d_out;
  float* post   = melout + (size_t)TMEL * MMEL;

  // ================= encoder =================
  embed_kernel<<<dim3(TPHN), dim3(EDIM), 0, stream>>>(src, emb, xA);
  float* curx = xA; float* nxtx = xB;
  for (int L = 0; L < 3; ++L) {            // 3x conv(k=5,pad=2)+BN+ReLU as tap-GEMMs
    cvt(curx, xpad, TPHN, EDIM, TPHN + 4, EDIM, 2);
    wtap(cW[L], BtC, EDIM, EDIM, 5, EDIM, EDIM);
    fold(cB[L], cS[L], cSh[L], EDIM);
    conv_taps(xpad, EDIM, 2, 2, 5, BtC, EDIM, EDIM, nxtx, EDIM, cS[L], fbuf, TPHN, EDIM, EDIM, 1);
    float* t = curx; curx = nxtx; nxtx = t;
  }
  cvt(curx, xpad, TPHN, EDIM, TPHN, EDIM, 0);
  wtap(efWih, BtG, 3 * HENC, EDIM, 1, EDIM, 3 * HENC);
  gemm(xpad, EDIM, BtG, 3 * HENC, giF, 3 * HENC, nullptr, efbih, TPHN, 3 * HENC, EDIM, 0, 2);
  wtap(ebWih, BtG, 3 * HENC, EDIM, 1, EDIM, 3 * HENC);
  gemm(xpad, EDIM, BtG, 3 * HENC, giB, 3 * HENC, nullptr, ebbih, TPHN, 3 * HENC, EDIM, 0, 2);
  gru_bidir_kernel<<<dim3(2), dim3(3 * HENC), 0, stream>>>(giF, giB, efWhh, ebWhh, efbhh, ebbhh,
                                                           enc, TPHN, HENC, 2 * HENC);

  // ================= duration expansion + decoder =================
  cumsum_kernel<<<dim3(1), dim3(1), 0, stream>>>(dur, csum, TPHN);
  pool_memory_kernel<<<dim3(CNT), dim3(EDIM), 0, stream>>>(csum, enc, mp);
  cvt(mp, mph, CNT, EDIM, CNT, EDIM, 0);
  wslice_f16_kernel<<<dim3((256 * 1536 + 255) / 256), dim3(256), 0, stream>>>(d0Wih, BtD);
  gemm(mph, EDIM, BtD, 3 * DDEC, gimem, 3 * DDEC, nullptr, d0bih, CNT, 3 * DDEC, EDIM, 0, 2);
  decoder_kernel<<<dim3(1), dim3(DDEC), 0, stream>>>(gimem, d0Wih, d0Whh, d0bhh,
                                                     d1Wih, d1Whh, d1bih, d1bhh,
                                                     pW1, pb1, pW2, pb2, mlW, mlb, melout, CNT);

  // ================= CBHG postnet =================
  cvt(melout, melpad, TMEL, MMEL, TMEL + 8, 96, 4);
  for (int k = 1; k <= 8; ++k) {           // conv banks -> ycat columns (k-1)*80
    const float* W  = (const float*)d_in[37 + 4 * (k - 1)];
    const float* b  = (const float*)d_in[38 + 4 * (k - 1)];
    const float* sc = (const float*)d_in[39 + 4 * (k - 1)];
    const float* sp = (const float*)d_in[40 + 4 * (k - 1)];
    wtap(W, BtBank, MMEL, MMEL, k, 96, MMEL);
    fold(b, sc, sp, MMEL);
    conv_taps(melpad, 96, 4, k / 2, k, BtBank, 96, MMEL,
              ycat + (size_t)(k - 1) * MMEL, 640, sc, fbuf, TMEL, MMEL, 96, 1);
  }
  { size_t tot = (size_t)(TMEL + 2) * 640;  // fused maxpool-over-time + f16 cvt
    maxpool_cvt_kernel<<<dim3((unsigned)((tot + 255) / 256)), dim3(256), 0, stream>>>(ycat, ypool, TMEL, 640); }
  wtap(q1W, BtP1, 256, 640, 3, 640, 256);
  fold(q1b, q1s, q1h, 256);
  conv_taps(ypool, 640, 1, 1, 3, BtP1, 640, 256, p1b, 256, q1s, fbuf, TMEL, 256, 640, 1);
  cvt(p1b, p1pad, TMEL, 256, TMEL + 2, 256, 1);
  wtap(q2W, BtP2, MMEL, 256, 3, 256, MMEL);
  fold(q2b, q2s, q2h, MMEL);
  conv_taps(p1pad, 256, 1, 1, 3, BtP2, 256, MMEL, hwy, MMEL, q2s, fbuf, TMEL, MMEL, 256, 0);
  addv(melout, hwy, (long)TMEL * MMEL);     // residual add
  for (int i = 0; i < 4; ++i) {            // highway x4
    const float* Wh = (const float*)d_in[77 + 4 * i];
    const float* bh = (const float*)d_in[78 + 4 * i];
    const float* Wt = (const float*)d_in[79 + 4 * i];
    const float* bt = (const float*)d_in[80 + 4 * i];
    cvt(hwy, hwypad, TMEL, MMEL, TMEL, 96, 0);
    wtap(Wh, BtHW, MMEL, MMEL, 1, 96, MMEL);
    gemm(hwypad, 96, BtHW, MMEL, hh, MMEL, nullptr, bh, TMEL, MMEL, 96, 1, 2);  // relu
    wtap(Wt, BtHW, MMEL, MMEL, 1, 96, MMEL);
    gemm(hwypad, 96, BtHW, MMEL, ht, MMEL, nullptr, bt, TMEL, MMEL, 96, 2, 2);  // sigmoid
    highway_kernel<<<dim3((unsigned)(((long)TMEL * MMEL + 255) / 256)), dim3(256), 0, stream>>>(hh, ht, hwy, (long)TMEL * MMEL);
  }
  cvt(hwy, hwypad, TMEL, MMEL, TMEL, 96, 0);
  wtap(gfWih, BtCG, 240, MMEL, 1, 96, 240);
  gemm(hwypad, 96, BtCG, 240, gicf, 240, nullptr, gfbih, TMEL, 240, 96, 0, 2);
  wtap(gbWih, BtCG, 240, MMEL, 1, 96, 240);
  gemm(hwypad, 96, BtCG, 240, gicb, 240, nullptr, gbbih, TMEL, 240, 96, 0, 2);
  gru_bidir_kernel<<<dim3(2), dim3(240), 0, stream>>>(gicf, gicb, gfWhh, gbWhh, gfbhh, gbbhh,
                                                      cb, TMEL, MMEL, 160);
  cvt(cb, cbh, TMEL, 160, TMEL, 160, 0);
  wtap(llW, BtLL, MMEL, 160, 1, 160, MMEL);
  gemm(cbh, 160, BtLL, MMEL, post, MMEL, nullptr, llb, TMEL, MMEL, 160, 0, 2);
  addv(melout, post, (long)TMEL * MMEL);    // mel_postnet = mel + residual
}
